// PoincareEmbHingeLoss_37684043055383
// MI455X (gfx1250) — matrix operand order repaired and checked by reading, not compiled, not run on hardware
//
#include <hip/hip_runtime.h>
#include <hip/hip_bf16.h>

typedef __attribute__((ext_vector_type(16))) __bf16 v16bf;
typedef __attribute__((ext_vector_type(8)))  float  v8f;

#define HL_MARGIN 0.1f
#define HL_BEPS   (1.0f - 1e-5f)
#define HL_AEPS   1e-7f
#define HL_B      4096
#define HL_C      10000
#define HL_D      64

__device__ __forceinline__ float arccosh_ref(float x) {
    x = fmaxf(x, 1.0f + HL_AEPS);
    return __logf(x + __fsqrt_rn(x * x - 1.0f));
}

// ---------------------------------------------------------------- init / finalize
__global__ void init_accum_kernel(float* accum) {
    if (threadIdx.x == 0) accum[0] = 0.0f;
}

__global__ void finalize_kernel(const float* accum, float* out) {
    if (threadIdx.x == 0)
        out[0] = (accum[0] - HL_MARGIN * (float)HL_B) / (float)HL_B;
}

// ---------------------------------------------------------------- per-row stats
__global__ void pred_stats_kernel(const float* __restrict__ pred,
                                  const float* __restrict__ targ,
                                  float* __restrict__ pn, float* __restrict__ ipn,
                                  float* __restrict__ d2c, int B) {
    int i = blockIdx.x * blockDim.x + threadIdx.x;
    if (i >= B) return;
    const float4* p = (const float4*)(pred + i * HL_D);
    const float4* t = (const float4*)(targ + i * HL_D);
    float sp = 0.0f, st = 0.0f, sd = 0.0f;
#pragma unroll
    for (int k = 0; k < HL_D / 4; ++k) {
        float4 a = p[k], b = t[k];
        sp += a.x * a.x + a.y * a.y + a.z * a.z + a.w * a.w;
        st += b.x * b.x + b.y * b.y + b.z * b.z + b.w * b.w;
        float dx = a.x - b.x, dy = a.y - b.y, dz = a.z - b.z, dw = a.w - b.w;
        sd += dx * dx + dy * dy + dz * dz + dw * dw;
    }
    float pnv = fminf(sp, HL_BEPS);
    float tnv = fminf(st, HL_BEPS);
    pn[i]  = pnv;
    ipn[i] = 1.0f / (1.0f - pnv);
    float x = 1.0f + 2.0f * sd / ((1.0f - pnv) * (1.0f - tnv));
    d2c[i] = arccosh_ref(x);
}

__global__ void all_stats_kernel(const float* __restrict__ alle,
                                 float* __restrict__ an, float* __restrict__ ian, int C) {
    int i = blockIdx.x * blockDim.x + threadIdx.x;
    if (i >= C) return;
    const float4* p = (const float4*)(alle + i * HL_D);
    float s = 0.0f;
#pragma unroll
    for (int k = 0; k < HL_D / 4; ++k) {
        float4 a = p[k];
        s += a.x * a.x + a.y * a.y + a.z * a.z + a.w * a.w;
    }
    float anv = fminf(s, HL_BEPS);
    an[i]  = anv;
    ian[i] = 1.0f / (1.0f - anv);
}

// ---------------------------------------------------------------- bf16 hi/lo fragment packing
// Layout: dword index u = (tile*32 + lane)*16 + jj ; jj in [0,16): jj>>3 = K-chunk (0..1),
// lane>>4 = half. A-role K map (ISA 16-bit A 16x32): k0 = chunk*32 + g*16 + half*8 + j4*2.
// B-role K map (ISA 16-bit B 32x16):                 k0 = chunk*32 + g*16 + half*8 + j4*2.
__global__ void pack_bf16_kernel(const float* __restrict__ X,
                                 unsigned* __restrict__ hiOut,
                                 unsigned* __restrict__ loOut,
                                 int total, int isB) {
    int u = blockIdx.x * blockDim.x + threadIdx.x;
    if (u >= total) return;
    int t = u >> 9;            // 512 dwords per 16-row tile
    int rem = u & 511;
    int l = rem >> 4;          // lane 0..31
    int jj = rem & 15;
    int half = l >> 4;
    int row = t * 16 + (l & 15);
    int chunk = jj >> 3;
    int j = jj & 7;
    int k0 = isB ? (chunk * 32 + ((j >> 2) << 4) + half * 8 + (j & 3) * 2)
                 : (chunk * 32 + ((j >> 2) << 4) + half * 8 + (j & 3) * 2);
    float x0 = X[row * HL_D + k0];
    float x1 = X[row * HL_D + k0 + 1];
    unsigned b0 = __float_as_uint(x0), b1 = __float_as_uint(x1);
    unsigned h0 = b0 & 0xFFFF0000u,  h1 = b1 & 0xFFFF0000u;
    float l0 = x0 - __uint_as_float(h0);
    float l1 = x1 - __uint_as_float(h1);
    hiOut[u] = (h0 >> 16) | (h1 & 0xFFFF0000u);
    loOut[u] = (__float_as_uint(l0) >> 16) | (__float_as_uint(l1) & 0xFFFF0000u);
}

// ---------------------------------------------------------------- WMMA GEMM + hinge epilogue
union Frag { uint4 q[2]; v16bf v; };

__device__ __forceinline__ v8f wmma_bf16(const v16bf& a, const v16bf& b, v8f c) {
    return __builtin_amdgcn_wmma_f32_16x16x32_bf16(false, a, false, b, (short)0, c, false, false);
}

__global__ __launch_bounds__(256) void gemm_hinge_kernel(
    const unsigned* __restrict__ pAHi, const unsigned* __restrict__ pALo,
    const unsigned* __restrict__ pBHi, const unsigned* __restrict__ pBLo,
    const float* __restrict__ pn, const float* __restrict__ ipn,
    const float* __restrict__ d2c,
    const float* __restrict__ an, const float* __restrict__ ian,
    float* __restrict__ accum, int nTiles, int nStrips) {
    const int lane = threadIdx.x & 31;
    const int wave = threadIdx.x >> 5;
    const int w = blockIdx.x * 8 + wave;
    const int mTile = w / nStrips;
    const int strip = w % nStrips;
    const int half = lane >> 4;

    // A fragments (hi/lo planes, 2 K-chunks each), reused across 4 n-tiles
    Frag aHi[2], aLo[2];
    {
        const uint4* bh = (const uint4*)(pAHi + (((size_t)mTile * 32 + lane) << 4));
        const uint4* bl = (const uint4*)(pALo + (((size_t)mTile * 32 + lane) << 4));
        aHi[0].q[0] = bh[0]; aHi[0].q[1] = bh[1];
        aHi[1].q[0] = bh[2]; aHi[1].q[1] = bh[3];
        aLo[0].q[0] = bl[0]; aLo[0].q[1] = bl[1];
        aLo[1].q[0] = bl[2]; aLo[1].q[1] = bl[3];
    }

    float pnv[8], ipnv[8], dcv[8];
    const int mBase = mTile * 16 + half * 8;
#pragma unroll
    for (int r = 0; r < 8; ++r) {
        pnv[r]  = pn[mBase + r];
        ipnv[r] = ipn[mBase + r];
        dcv[r]  = d2c[mBase + r];
    }

    float hinge = 0.0f;
    const int n0 = strip * 4;
    const int nEnd = (n0 + 4 < nTiles) ? (n0 + 4) : nTiles;   // wave-uniform
    for (int nt = n0; nt < nEnd; ++nt) {
        Frag bHi[2], bLo[2];
        const uint4* bh = (const uint4*)(pBHi + (((size_t)nt * 32 + lane) << 4));
        const uint4* bl = (const uint4*)(pBLo + (((size_t)nt * 32 + lane) << 4));
        bHi[0].q[0] = bh[0]; bHi[0].q[1] = bh[1];
        bHi[1].q[0] = bh[2]; bHi[1].q[1] = bh[3];
        bLo[0].q[0] = bl[0]; bLo[0].q[1] = bl[1];
        bLo[1].q[0] = bl[2]; bLo[1].q[1] = bl[3];

        v8f c = {0.f, 0.f, 0.f, 0.f, 0.f, 0.f, 0.f, 0.f};
        // bf16x3 split: Ah*Bh + Ah*Bl + Al*Bh per K-chunk
        c = wmma_bf16(aHi[0].v, bHi[0].v, c);
        c = wmma_bf16(aHi[0].v, bLo[0].v, c);
        c = wmma_bf16(aLo[0].v, bHi[0].v, c);
        c = wmma_bf16(aHi[1].v, bHi[1].v, c);
        c = wmma_bf16(aHi[1].v, bLo[1].v, c);
        c = wmma_bf16(aLo[1].v, bHi[1].v, c);

        const int n = nt * 16 + (lane & 15);
        const float anv = an[n], ianv = ian[n];
#pragma unroll
        for (int r = 0; r < 8; ++r) {
            float dot = c[r];
            float sqd = fmaxf(pnv[r] + anv - 2.0f * dot, 0.0f);
            float x = fmaf(2.0f * sqd, ipnv[r] * ianv, 1.0f);
            x = fmaxf(x, 1.0f + HL_AEPS);
            float d = __logf(x + __fsqrt_rn(x * x - 1.0f));
            hinge += fmaxf(dcv[r] - d + HL_MARGIN, 0.0f);
        }
    }

    __shared__ float sred[256];
    sred[threadIdx.x] = hinge;
    __syncthreads();
    for (int s = 128; s > 0; s >>= 1) {
        if (threadIdx.x < s) sred[threadIdx.x] += sred[threadIdx.x + s];
        __syncthreads();
    }
    if (threadIdx.x == 0) atomicAdd(accum, sred[0]);
}

// ---------------------------------------------------------------- launch
extern "C" void kernel_launch(void* const* d_in, const int* in_sizes, int n_in,
                              void* d_out, int out_size, void* d_ws, size_t ws_size,
                              hipStream_t stream) {
    const float* pred = (const float*)d_in[0];   // [B,64]
    const float* targ = (const float*)d_in[1];   // [B,64]
    const float* alle = (const float*)d_in[2];   // [C,64]
    float* out = (float*)d_out;

    // workspace carve-out (dwords)
    float* accum = (float*)d_ws;
    float* pn  = accum + 64;
    float* ipn = pn + HL_B;
    float* d2c = ipn + HL_B;
    float* an  = d2c + HL_B;
    float* ian = an + 10016;
    unsigned* packAHi = (unsigned*)(ian + 10016);
    unsigned* packALo = packAHi + HL_B * 32;     // 131072 dwords each
    unsigned* packBHi = packALo + HL_B * 32;
    unsigned* packBLo = packBHi + HL_C * 32;     // 320000 dwords each

    init_accum_kernel<<<1, 64, 0, stream>>>(accum);

    pred_stats_kernel<<<(HL_B + 127) / 128, 128, 0, stream>>>(pred, targ, pn, ipn, d2c, HL_B);
    all_stats_kernel<<<(HL_C + 127) / 128, 128, 0, stream>>>(alle, an, ian, HL_C);

    const int totA = HL_B * 32;                  // dwords per plane
    const int totB = HL_C * 32;
    pack_bf16_kernel<<<(totA + 255) / 256, 256, 0, stream>>>(pred, packAHi, packALo, totA, 0);
    pack_bf16_kernel<<<(totB + 255) / 256, 256, 0, stream>>>(alle, packBHi, packBLo, totB, 1);

    const int nTiles  = HL_C / 16;               // 625
    const int nStrips = (nTiles + 3) / 4;        // 157
    const int mTiles  = HL_B / 16;               // 256
    const int waves   = mTiles * nStrips;        // 40192
    gemm_hinge_kernel<<<waves / 8, 256, 0, stream>>>(
        packAHi, packALo, packBHi, packBLo, pn, ipn, d2c, an, ian, accum, nTiles, nStrips);

    finalize_kernel<<<1, 64, 0, stream>>>(accum, out);
}
